// SimpleGradientField_45707041964604
// MI455X (gfx1250) — matrix operand order repaired and checked by reading, not compiled, not run on hardware
//
#include <hip/hip_runtime.h>

typedef __attribute__((ext_vector_type(16))) _Float16 v16h;
typedef __attribute__((ext_vector_type(8)))  float    v8f;

#define N_PARTS 512
#define BATCHES 8
#define EPSV    1e-6f

// workspace layout (float offsets)
#define WS_C    0                 // [8][32]  c_b = W2 @ ts
#define WS_S0   256               // [8]      s0 = b2 . ts
#define WS_SSC  264               // [8]      self scalar
#define WS_SG   272               // [8]      self g
#define WS_PF   280               // [8][512] per-block first partial
#define WS_PS   (280 + 4096)      // [8][512] per-block second partial
#define WS_V    (280 + 8192)      // [8][512][3] per-particle v sums / 511

// ---- CDNA5 hardware transcendentals -----------------------------------------
__device__ __forceinline__ float fast_tanh(float x) {
#if __has_builtin(__builtin_amdgcn_tanhf)
  return __builtin_amdgcn_tanhf(x);          // v_tanh_f32 (TRANS32)
#else
  float r;
  // raw v_tanh_f32; v_nops cover the TRANS result-use hazard
  asm volatile("v_tanh_f32 %0, %1\n\tv_nop\n\tv_nop" : "=v"(r) : "v"(x));
  return r;
#endif
}

__device__ __forceinline__ float fast_sqrt(float x) {
#if __has_builtin(__builtin_amdgcn_sqrtf)
  return __builtin_amdgcn_sqrtf(x);          // v_sqrt_f32, input >= 1e-6 here
#else
  return sqrtf(x);
#endif
}

template <int MASK>
__device__ __forceinline__ float xor_add(float v) {
  // ds_swizzle_b32 group-of-32: and=0x1F, or=0, xor=MASK
  int x = __builtin_bit_cast(int, v);
  int y = __builtin_amdgcn_ds_swizzle(x, 0x7C00 | MASK);
  return v + __builtin_bit_cast(float, y);
}

__device__ __forceinline__ float wave_sum32(float v) {
  v = xor_add<1>(v); v = xor_add<2>(v); v = xor_add<4>(v);
  v = xor_add<8>(v); v = xor_add<16>(v);
  return v;
}

// ---------------- prolog: per-batch constants + self-pair terms ----------------
__global__ void sgf_prolog(const float* __restrict__ t,
                           const float* __restrict__ Wse, const float* __restrict__ bse,
                           const float* __restrict__ W1,  const float* __restrict__ b1,
                           const float* __restrict__ W2,  const float* __restrict__ b2,
                           const float* __restrict__ Wt,  const float* __restrict__ bt,
                           float* __restrict__ ws) {
  const int b = threadIdx.x;
  if (b >= BATCHES) return;
  const float tb = t[b];
  float ts[32];
  for (int k = 0; k < 32; ++k) ts[k] = fmaf(tb, Wt[k], bt[k]);
  float s0 = 0.f;
  for (int k = 0; k < 32; ++k) s0 = fmaf(b2[k], ts[k], s0);
  float c[32];
  for (int f = 0; f < 32; ++f) {
    float a = 0.f;
    for (int k = 0; k < 32; ++k) a = fmaf(W2[f * 32 + k], ts[k], a);
    c[f] = a;
    ws[WS_C + b * 32 + f] = a;
  }
  ws[WS_S0 + b] = s0;

  // self pair: d = sqrt(eps)
  const float d0 = fast_sqrt(EPSV);
  float h1[32], ap[32];
  for (int h = 0; h < 32; ++h) {
    const float th = fast_tanh(fmaf(d0, Wse[h], bse[h]));
    h1[h] = th;
    ap[h] = d0 * Wse[h] * (1.f - th * th);
  }
  float sscal = s0, g = 0.f, e[32];
  for (int n = 0; n < 32; ++n) {
    float z = b1[n];
    for (int h = 0; h < 32; ++h) z = fmaf(h1[h], W1[h * 32 + n], z);
    const float h2 = fast_tanh(z);
    sscal = fmaf(h2, c[n], sscal);
    e[n] = (1.f - h2 * h2) * c[n];
  }
  for (int n = 0; n < 32; ++n) {
    float q = 0.f;
    for (int h = 0; h < 32; ++h) q = fmaf(ap[h], W1[h * 32 + n], q);
    g = fmaf(q, e[n], g);
  }
  ws[WS_SSC + b] = sscal;
  ws[WS_SG + b]  = g;
}

// ---------------- main: one block per (particle i, batch b) ----------------
__global__ __launch_bounds__(256) void sgf_main(const float* __restrict__ x,
                                                const float* __restrict__ Wse,
                                                const float* __restrict__ bse,
                                                const float* __restrict__ W1,
                                                const float* __restrict__ b1,
                                                float* __restrict__ ws) {
  const int i    = blockIdx.x;
  const int b    = blockIdx.y;
  const int tid  = threadIdx.x;
  const int wv   = tid >> 5;
  const int lane = tid & 31;
  const int ln   = lane & 15;        // column / row-in-tile index
  const int hi   = lane >> 4;        // lane half
  const int hi8  = hi << 3;

  __shared__ float scal[8][16];
  __shared__ float part[8][5];

  // per-lane K set of the f16 A fragment: k = e + (e&8) + 8*hi
  float wsev[16], bsev[16];
#pragma unroll
  for (int e = 0; e < 16; ++e) {
    const int k = e + (e & 8) + hi8;
    wsev[e] = Wse[k];
    bsev[e] = bse[k];
  }
  // B fragments of W1 (32x16 each): lane ln = column, K = e + (e&8) + 8*hi
  v16h bf_lo, bf_hi;
#pragma unroll
  for (int e = 0; e < 16; ++e) {
    const int k = e + (e & 8) + (hi << 3);
    bf_lo[e] = (_Float16)W1[k * 32 + ln];
    bf_hi[e] = (_Float16)W1[k * 32 + 16 + ln];
  }
  const float b1lo = b1[ln], b1hi = b1[16 + ln];
  const float c0 = ws[WS_C + b * 32 + ln];
  const float c1 = ws[WS_C + b * 32 + 16 + ln];
  const float s0 = ws[WS_S0 + b];

  const float* __restrict__ xb = x + (size_t)b * (N_PARTS * 3);
  const float xi0 = xb[i * 3 + 0], xi1 = xb[i * 3 + 1], xi2 = xb[i * 3 + 2];

  float facc = 0.f, hacc = 0.f;
  float vx = 0.f, vy = 0.f, vz = 0.f;

  for (int tt = 0; tt < 4; ++tt) {
    const int j = (((wv << 2) + tt) << 4) + ln;       // pair row index
    const float rx = xi0 - xb[j * 3 + 0];
    const float ry = xi1 - xb[j * 3 + 1];
    const float rz = xi2 - xb[j * 3 + 2];
    const float d  = fast_sqrt(fmaf(rx, rx, fmaf(ry, ry, fmaf(rz, rz, EPSV))));

    // forward A = h1 (tanh layer 1), backward A' = d * Wse * (1 - h1^2)
    v16h afrag, apfrag;
#pragma unroll
    for (int e = 0; e < 16; ++e) {
      const float h = fast_tanh(fmaf(d, wsev[e], bsev[e]));
      afrag[e]  = (_Float16)h;
      apfrag[e] = (_Float16)(d * wsev[e] * (1.f - h * h));
    }

    v8f acc0, acc1, q0, q1;
#pragma unroll
    for (int r = 0; r < 8; ++r) { acc0[r] = b1lo; acc1[r] = b1hi; q0[r] = 0.f; q1[r] = 0.f; }
    acc0 = __builtin_amdgcn_wmma_f32_16x16x32_f16(false, afrag,  false, bf_lo, (short)0, acc0, false, false);
    acc1 = __builtin_amdgcn_wmma_f32_16x16x32_f16(false, afrag,  false, bf_hi, (short)0, acc1, false, false);
    q0   = __builtin_amdgcn_wmma_f32_16x16x32_f16(false, apfrag, false, bf_lo, (short)0, q0,   false, false);
    q1   = __builtin_amdgcn_wmma_f32_16x16x32_f16(false, apfrag, false, bf_hi, (short)0, q1,   false, false);

    float srow[8];
#pragma unroll
    for (int r = 0; r < 8; ++r) {
      const float t0 = fast_tanh(acc0[r]);
      const float t1 = fast_tanh(acc1[r]);
      const float e0 = (1.f - t0 * t0) * c0;
      const float e1 = (1.f - t1 * t1) * c1;
      facc += q0[r] * e0 + q1[r] * e1;     // g partials (d already folded into A')
      const float s = fmaf(t0, c0, t1 * c1);
      hacc += s;                            // h2.c partials for "second"
      srow[r] = s;
    }
    // reduce over the 16 columns (lanes within each half)
#pragma unroll
    for (int r = 0; r < 8; ++r) {
      srow[r] = xor_add<1>(srow[r]);
      srow[r] = xor_add<2>(srow[r]);
      srow[r] = xor_add<4>(srow[r]);
      srow[r] = xor_add<8>(srow[r]);
    }
    if (ln == 0) {                          // lanes 0 and 16 hold rows 0-7 / 8-15
#pragma unroll
      for (int r = 0; r < 8; ++r) scal[wv][hi8 + r] = srow[r] + s0;
    }
    const float sc = scal[wv][ln];          // same-wave LDS ops are in-order
    if (hi == 0) {                          // one lane per pair accumulates v
      vx = fmaf(rx, sc, vx);
      vy = fmaf(ry, sc, vy);
      vz = fmaf(rz, sc, vz);
    }
  }

  facc = wave_sum32(facc);
  hacc = wave_sum32(hacc);
  vx = wave_sum32(vx); vy = wave_sum32(vy); vz = wave_sum32(vz);
  if (lane == 0) {
    part[wv][0] = facc; part[wv][1] = hacc;
    part[wv][2] = vx;   part[wv][3] = vy;   part[wv][4] = vz;
  }
  __syncthreads();
  if (tid == 0) {
    float F = 0.f, H = 0.f, VX = 0.f, VY = 0.f, VZ = 0.f;
    for (int w = 0; w < 8; ++w) {
      F += part[w][0]; H += part[w][1];
      VX += part[w][2]; VY += part[w][3]; VZ += part[w][4];
    }
    ws[WS_PF + b * N_PARTS + i] = F;
    ws[WS_PS + b * N_PARTS + i] = 3.f * (H + 512.f * s0);
    const float inv = 1.f / 511.f;
    const size_t vo = WS_V + ((size_t)b * N_PARTS + i) * 3;
    ws[vo + 0] = VX * inv; ws[vo + 1] = VY * inv; ws[vo + 2] = VZ * inv;
  }
}

// ---------------- finalize: COM removal + trace ----------------
__global__ __launch_bounds__(512) void sgf_final(const float* __restrict__ ws,
                                                 float* __restrict__ out) {
  const int b = blockIdx.x;
  const int i = threadIdx.x;
  const int wv = i >> 5, lane = i & 31;
  __shared__ float part[16][5];
  __shared__ float tot[5];

  const size_t vo = WS_V + ((size_t)b * N_PARTS + i) * 3;
  const float v0 = ws[vo + 0], v1 = ws[vo + 1], v2 = ws[vo + 2];
  const float pf = ws[WS_PF + b * N_PARTS + i];
  const float ps = ws[WS_PS + b * N_PARTS + i];

  float r0 = wave_sum32(v0), r1 = wave_sum32(v1), r2 = wave_sum32(v2);
  float r3 = wave_sum32(pf), r4 = wave_sum32(ps);
  if (lane == 0) {
    part[wv][0] = r0; part[wv][1] = r1; part[wv][2] = r2;
    part[wv][3] = r3; part[wv][4] = r4;
  }
  __syncthreads();
  if (i == 0) {
    float a0 = 0.f, a1 = 0.f, a2 = 0.f, a3 = 0.f, a4 = 0.f;
    for (int w = 0; w < 16; ++w) {
      a0 += part[w][0]; a1 += part[w][1]; a2 += part[w][2];
      a3 += part[w][3]; a4 += part[w][4];
    }
    tot[0] = a0; tot[1] = a1; tot[2] = a2; tot[3] = a3; tot[4] = a4;
  }
  __syncthreads();
  const float m0 = tot[0] * (1.f / 512.f);
  const float m1 = tot[1] * (1.f / 512.f);
  const float m2 = tot[2] * (1.f / 512.f);
  out[(size_t)b * 1536 + i * 3 + 0] = v0 - m0;
  out[(size_t)b * 1536 + i * 3 + 1] = v1 - m1;
  out[(size_t)b * 1536 + i * 3 + 2] = v2 - m2;
  if (i == 0) {
    const float first  = tot[3] - 512.f  * ws[WS_SG + b];   // remove self pairs
    const float second = tot[4] - 1536.f * ws[WS_SSC + b];  // 3 * 512 * self scalar
    out[12288 + b] = (first + second) * (1.f / 511.f);
  }
}

extern "C" void kernel_launch(void* const* d_in, const int* in_sizes, int n_in,
                              void* d_out, int out_size, void* d_ws, size_t ws_size,
                              hipStream_t stream) {
  (void)in_sizes; (void)n_in; (void)out_size; (void)ws_size;
  const float* x   = (const float*)d_in[0];
  const float* t   = (const float*)d_in[1];
  const float* Wse = (const float*)d_in[2];
  const float* bse = (const float*)d_in[3];
  const float* W1  = (const float*)d_in[4];
  const float* b1  = (const float*)d_in[5];
  const float* W2  = (const float*)d_in[6];
  const float* b2  = (const float*)d_in[7];
  const float* Wt  = (const float*)d_in[8];
  const float* bt  = (const float*)d_in[9];
  float* ws  = (float*)d_ws;
  float* out = (float*)d_out;

  sgf_prolog<<<1, 64, 0, stream>>>(t, Wse, bse, W1, b1, W2, b2, Wt, bt, ws);
  sgf_main<<<dim3(N_PARTS, BATCHES), 256, 0, stream>>>(x, Wse, bse, W1, b1, ws);
  sgf_final<<<BATCHES, N_PARTS, 0, stream>>>(ws, out);
}
